// OnnxGPT_7662221656207
// MI455X (gfx1250) — matrix operand order repaired and checked by reading, not compiled, not run on hardware
//
#include <hip/hip_runtime.h>
#include <hip/hip_bf16.h>
#include <math.h>

// ---------------- model constants ----------------
constexpr int Lx    = 12;
constexpr int NH    = 16;
constexpr int NKV   = 4;
constexpr int HD    = 64;
constexpr int Cx    = NH * HD;      // 1024
constexpr int KVx   = NKV * HD;     // 256
constexpr int FFx   = 4 * Cx;       // 4096
constexpr int Tx    = 1024;
constexpr int VOCAB = 50257;
constexpr int VPAD  = 50304;

// ---------------- WMMA types ----------------
typedef __attribute__((ext_vector_type(16))) __bf16 v16bf;
typedef __attribute__((ext_vector_type(8)))  __bf16 v8bf;
typedef __attribute__((ext_vector_type(4)))  __bf16 v4bf;
typedef __attribute__((ext_vector_type(2)))  __bf16 v2bf;
typedef __attribute__((ext_vector_type(8)))  float  v8f;

// native fptrunc -> lets the backend pick the hardware bf16 convert if present
__device__ __forceinline__ __bf16 f2bf(float f) { return (__bf16)f; }
__device__ __forceinline__ v2bf f2bf2(float lo, float hi) {
#if __has_builtin(__builtin_amdgcn_cvt_pk_bf16_f32)
    return __builtin_amdgcn_cvt_pk_bf16_f32(lo, hi);
#else
    v2bf r; r[0] = (__bf16)lo; r[1] = (__bf16)hi; return r;
#endif
}
__device__ __forceinline__ float4 ld4(const float* p) { return *(const float4*)p; }
__device__ __forceinline__ v4bf cvt4(float4 a) {
    return __builtin_shufflevector(f2bf2(a.x, a.y), f2bf2(a.z, a.w), 0, 1, 2, 3);
}
__device__ __forceinline__ v8bf cvt8(float4 a, float4 b) {
    return __builtin_shufflevector(cvt4(a), cvt4(b), 0, 1, 2, 3, 4, 5, 6, 7);
}
__device__ __forceinline__ v16bf concat8(v8bf lo, v8bf hi) {
    return __builtin_shufflevector(lo, hi, 0,1,2,3,4,5,6,7,8,9,10,11,12,13,14,15);
}

// =====================================================================
// Generic GEMM: C[M x N] = epilogue(A[M x K] * B[K x N])
// fp32 in global -> bf16 in LDS (A row-major, B K-packed per column),
// double-buffered; fragments are 2 x ds_load_b128.
// Block = 256 thr (8 waves); tile 64x128; wave = 16x64 (4 WMMA accum).
// flag: 0=store, 1=add to C, 2=relu^2, 3=15*tanh(x/15). ncap bounds cols.
// =====================================================================
#define GBM 64
#define GBN 128
#define GKT 32
#define GKP (GKT + 8)   // 40 elems = 80B pitch (16B multiple)

__global__ __launch_bounds__(256) void gemm_bf16_kernel(
    const float* __restrict__ A, const float* __restrict__ B,
    float* __restrict__ Cc, int M, int N, int K, int ldc, int ncap, int flag)
{
    __shared__ __bf16 As[2][GBM][GKP];
    __shared__ __bf16 Bp[2][GBN][GKP];   // Bp[col][k]

    const int tid  = threadIdx.x;
    const int lane = tid & 31;
    const int w    = tid >> 5;
    const int m0   = blockIdx.y * GBM;
    const int n0   = blockIdx.x * GBN;
    const int mrowL = (w & 3) * 16;
    const int ncolL = (w >> 2) * 64;
    const int l15  = lane & 15;
    const int half = lane >> 4;
    const int khb  = half * 8;    // A frag K base
    const int kbb  = half * 16;   // unused; B frags use khb too

    // staging maps
    const int aR0 = tid >> 3,  aC0 = (tid & 7) * 4;          // A: i=0 -> rows 0..31
    const int bCg = (tid & 31) * 4, bRg = (tid >> 5) * 4;    // B: 4 cols x 4 rows

    v8f acc[4] = {};
    float4 aR[2];
    float  bRa[4][4];

    const int nk = K / GKT;

    // ---- prologue: load + store tile 0 ----
    {
        aR[0] = ld4(A + (size_t)(m0 + aR0) * K + aC0);
        aR[1] = ld4(A + (size_t)(m0 + aR0 + 32) * K + aC0);
        #pragma unroll
        for (int i = 0; i < 4; ++i) {
            float4 t = ld4(B + (size_t)(bRg + i) * N + n0 + bCg);
            bRa[i][0] = t.x; bRa[i][1] = t.y; bRa[i][2] = t.z; bRa[i][3] = t.w;
        }
        *(v4bf*)&As[0][aR0][aC0]      = cvt4(aR[0]);
        *(v4bf*)&As[0][aR0 + 32][aC0] = cvt4(aR[1]);
        #pragma unroll
        for (int j = 0; j < 4; ++j) {
            v4bf cv = __builtin_shufflevector(f2bf2(bRa[0][j], bRa[1][j]),
                                              f2bf2(bRa[2][j], bRa[3][j]), 0, 1, 2, 3);
            *(v4bf*)&Bp[0][bCg + j][bRg] = cv;
        }
    }

    for (int ks = 0; ks < nk; ++ks) {
        const int cur = ks & 1;
        __syncthreads();

        const bool more = (ks + 1 < nk);
        if (more) {
            const int kt = (ks + 1) * GKT;
            aR[0] = ld4(A + (size_t)(m0 + aR0) * K + kt + aC0);
            aR[1] = ld4(A + (size_t)(m0 + aR0 + 32) * K + kt + aC0);
            #pragma unroll
            for (int i = 0; i < 4; ++i) {
                float4 t = ld4(B + (size_t)(kt + bRg + i) * N + n0 + bCg);
                bRa[i][0] = t.x; bRa[i][1] = t.y; bRa[i][2] = t.z; bRa[i][3] = t.w;
            }
        }
        if (ks + 2 < nk) {
            __builtin_prefetch(A + (size_t)(m0 + (tid & 63)) * K + (ks + 2) * GKT, 0, 1);
            __builtin_prefetch(B + (size_t)((ks + 2) * GKT + (tid >> 3)) * N + n0 + ((tid & 7) << 4), 0, 1);
        }

        // ---- compute from LDS[cur]: preload frags, then WMMA clause ----
        const __bf16* ap = &As[cur][mrowL + l15][khb];
        v16bf af = concat8(*(const v8bf*)ap, *(const v8bf*)(ap + 16));
        v16bf bfr[4];
        #pragma unroll
        for (int bt = 0; bt < 4; ++bt) {
            const __bf16* bp = &Bp[cur][ncolL + bt * 16 + l15][khb];
            bfr[bt] = concat8(*(const v8bf*)bp, *(const v8bf*)(bp + 16));
        }
        #pragma unroll
        for (int bt = 0; bt < 4; ++bt)
            acc[bt] = __builtin_amdgcn_wmma_f32_16x16x32_bf16(
                false, af, false, bfr[bt], (short)0, acc[bt], false, false);

        if (more) {
            const int nxt = (ks + 1) & 1;
            *(v4bf*)&As[nxt][aR0][aC0]      = cvt4(aR[0]);
            *(v4bf*)&As[nxt][aR0 + 32][aC0] = cvt4(aR[1]);
            #pragma unroll
            for (int j = 0; j < 4; ++j) {
                v4bf cv = __builtin_shufflevector(f2bf2(bRa[0][j], bRa[1][j]),
                                                  f2bf2(bRa[2][j], bRa[3][j]), 0, 1, 2, 3);
                *(v4bf*)&Bp[nxt][bCg + j][bRg] = cv;
            }
        }
    }

    // ---- epilogue: D layout -> lane = col, vgpr r = row (+8 upper half) ----
    #pragma unroll
    for (int bt = 0; bt < 4; ++bt) {
        int col = n0 + ncolL + bt * 16 + l15;
        if (col >= ncap) continue;
        #pragma unroll
        for (int r = 0; r < 8; ++r) {
            int row = m0 + mrowL + r + 8 * half;
            float vv = acc[bt][r];
            if (flag == 1)      vv += Cc[(size_t)row * ldc + col];
            else if (flag == 2) { vv = vv > 0.f ? vv : 0.f; vv = vv * vv; }
            else if (flag == 3) { vv = 15.f * tanhf(vv * (1.f / 15.f)); }
            Cc[(size_t)row * ldc + col] = vv;
        }
    }
}

// =====================================================================
// Fused flash attention. Block = 256 thr (8 waves), one head, 128 queries.
// K block staged row-major (Ks[key][d]), V staged transposed (Vt[d][key])
// so S and P.V fragments are contiguous b128 LDS loads. Online softmax.
// Q pre-scaled by 1.2/sqrt(HD), K by 1.2.
// =====================================================================
__global__ __launch_bounds__(256) void attn_kernel(
    const float* __restrict__ qh, const float* __restrict__ kmat,
    const float* __restrict__ vmat, float* __restrict__ y)
{
    __shared__ __bf16 Ks[32][72];        // 144B pitch
    __shared__ __bf16 Vt[64][40];        // 80B pitch
    __shared__ __bf16 Pl[8][16][40];     // per-wave P staging

    const int tid  = threadIdx.x;
    const int lane = tid & 31;
    const int w    = tid >> 5;
    const int head = blockIdx.x;
    const int qb   = blockIdx.y * 8 + w;
    const int kvh  = head >> 2;          // NH/NKV = 4
    const int half = lane >> 4;
    const int l15  = lane & 15;
    const int khb  = half * 8;
    const int kbb  = half * 16;

    // Q A-fragments (K-dim = HD, halves 0..31 / 32..63), wide f32 loads
    const float* qrow = qh + (size_t)(qb * 16 + l15) * Cx + head * HD;
    v16bf aq0 = concat8(cvt8(ld4(qrow + khb),      ld4(qrow + khb + 4)),
                        cvt8(ld4(qrow + khb + 16), ld4(qrow + khb + 20)));
    v16bf aq1 = concat8(cvt8(ld4(qrow + 32 + khb),      ld4(qrow + 32 + khb + 4)),
                        cvt8(ld4(qrow + 32 + khb + 16), ld4(qrow + 32 + khb + 20)));

    float mrow[8], lrow[8];
    #pragma unroll
    for (int r = 0; r < 8; ++r) { mrow[r] = -3.0e38f; lrow[r] = 0.f; }
    v8f oc[4] = {};

    const int qmax      = qb * 16 + 15;
    const int blockQmax = blockIdx.y * 128 + 127;

    for (int kb = 0; kb <= blockQmax; kb += 32) {
        // ---- stage K (row-major) and V (transposed), 32 keys x 64 d ----
        #pragma unroll
        for (int i = 0; i < 2; ++i) {
            int idx4 = tid + i * 256;                 // 512 float4
            int key = idx4 >> 4, c4 = (idx4 & 15) * 4;
            float4 tk = ld4(kmat + (size_t)(kb + key) * KVx + kvh * HD + c4);
            *(v4bf*)&Ks[key][c4] = cvt4(tk);
            float4 tv = ld4(vmat + (size_t)(kb + key) * KVx + kvh * HD + c4);
            v2bf p01 = f2bf2(tv.x, tv.y), p23 = f2bf2(tv.z, tv.w);
            Vt[c4 + 0][key] = p01[0];
            Vt[c4 + 1][key] = p01[1];
            Vt[c4 + 2][key] = p23[0];
            Vt[c4 + 3][key] = p23[1];
        }
        __syncthreads();

        if (kb <= qmax) {
            v8f s[2];
            #pragma unroll
            for (int tI = 0; tI < 2; ++tI) {
                const __bf16* kp = &Ks[tI * 16 + l15][0];
                v16bf bk0 = concat8(*(const v8bf*)(kp + khb),      *(const v8bf*)(kp + khb + 16));
                v16bf bk1 = concat8(*(const v8bf*)(kp + 32 + khb), *(const v8bf*)(kp + 32 + khb + 16));
                v8f sc = {};
                sc = __builtin_amdgcn_wmma_f32_16x16x32_bf16(false, aq0, false, bk0, (short)0, sc, false, false);
                sc = __builtin_amdgcn_wmma_f32_16x16x32_bf16(false, aq1, false, bk1, (short)0, sc, false, false);
                int key = kb + tI * 16 + l15;
                #pragma unroll
                for (int r = 0; r < 8; ++r) {
                    int qr = qb * 16 + r + 8 * half;
                    if (key > qr) sc[r] = -1.0e30f;
                }
                s[tI] = sc;
            }

            // online softmax per query row
            #pragma unroll
            for (int r = 0; r < 8; ++r) {
                float tmax = fmaxf(s[0][r], s[1][r]);
                #pragma unroll
                for (int m = 1; m <= 8; m <<= 1)
                    tmax = fmaxf(tmax, __shfl_xor(tmax, m, 32));
                float mnew = fmaxf(mrow[r], tmax);
                float corr = __expf(mrow[r] - mnew);
                float p0 = __expf(s[0][r] - mnew);
                float p1 = __expf(s[1][r] - mnew);
                s[0][r] = p0; s[1][r] = p1;
                float rs = p0 + p1;
                #pragma unroll
                for (int m = 1; m <= 8; m <<= 1)
                    rs += __shfl_xor(rs, m, 32);
                lrow[r] = lrow[r] * corr + rs;
                mrow[r] = mnew;
                #pragma unroll
                for (int dt = 0; dt < 4; ++dt) oc[dt][r] = oc[dt][r] * corr;
            }

            // P (D layout) -> per-wave LDS -> A-fragment
            #pragma unroll
            for (int r = 0; r < 8; ++r) {
                v2bf pp2 = f2bf2(s[0][r], s[1][r]);
                Pl[w][r + 8 * half][l15]      = pp2[0];
                Pl[w][r + 8 * half][16 + l15] = pp2[1];
            }
            asm volatile("s_wait_dscnt 0" ::: "memory");
            const __bf16* pp = &Pl[w][l15][khb];
            v16bf pa = concat8(*(const v8bf*)pp, *(const v8bf*)(pp + 16));

            // Y += P.V : preload all V fragments, then WMMA clause
            v16bf vb[4];
            #pragma unroll
            for (int dt = 0; dt < 4; ++dt) {
                const __bf16* vp = &Vt[dt * 16 + l15][khb];
                vb[dt] = concat8(*(const v8bf*)vp, *(const v8bf*)(vp + 16));
            }
            #pragma unroll
            for (int dt = 0; dt < 4; ++dt)
                oc[dt] = __builtin_amdgcn_wmma_f32_16x16x32_bf16(
                    false, pa, false, vb[dt], (short)0, oc[dt], false, false);
        }
        __syncthreads();
    }

    #pragma unroll
    for (int dt = 0; dt < 4; ++dt) {
        int d = dt * 16 + l15;
        #pragma unroll
        for (int r = 0; r < 8; ++r) {
            int t = qb * 16 + r + 8 * half;
            y[(size_t)t * Cx + head * HD + d] = oc[dt][r] / lrow[r];
        }
    }
}

// =====================================================================
// Elementwise / normalization kernels
// =====================================================================
__global__ __launch_bounds__(256) void embed_kernel(
    const int* __restrict__ ids, const float* __restrict__ wte,
    float* __restrict__ xe, float* __restrict__ prev_out)
{
    __shared__ float red[256];
    int t = blockIdx.x, tid = threadIdx.x;
    int id = ids[t];
    const float* wr = wte + (size_t)id * Cx;
    float v[4]; float ss = 0.f;
    #pragma unroll
    for (int i = 0; i < 4; ++i) { v[i] = wr[tid + i * 256]; ss += v[i] * v[i]; }
    red[tid] = ss; __syncthreads();
    for (int s = 128; s > 0; s >>= 1) { if (tid < s) red[tid] += red[tid + s]; __syncthreads(); }
    float inv = rsqrtf(red[0] / Cx + 1e-6f);
    #pragma unroll
    for (int i = 0; i < 4; ++i) {
        float o = v[i] * inv;
        xe[(size_t)t * Cx + tid + i * 256] = o;
        if (t == Tx - 1) prev_out[tid + i * 256] = o;
    }
}

__global__ __launch_bounds__(256) void smear_kernel(
    const float* __restrict__ xe, const float* __restrict__ sgw,
    const float* __restrict__ slam, float* __restrict__ x, float* __restrict__ x0)
{
    int t = blockIdx.x, tid = threadIdx.x;
    float gate = 0.f;
    if (t > 0) {
        float dot = 0.f;
        #pragma unroll
        for (int j = 0; j < 24; ++j) dot += xe[(size_t)t * Cx + j] * sgw[j];
        gate = slam[0] / (1.f + __expf(-dot));
    }
    #pragma unroll
    for (int i = 0; i < 4; ++i) {
        int c = tid + i * 256;
        float o = xe[(size_t)t * Cx + c];
        if (t > 0) o += gate * xe[(size_t)(t - 1) * Cx + c];
        x[(size_t)t * Cx + c] = o;
        x0[(size_t)t * Cx + c] = o;
    }
}

// x = rl*x + xls*xl*x0 ; xn = rms(x). Null rlp -> 1, null xlp -> 0.
__global__ __launch_bounds__(256) void resid_rms_kernel(
    float* __restrict__ x, const float* __restrict__ x0, float* __restrict__ xn,
    const float* __restrict__ rlp, const float* __restrict__ xlp, int idx, float xls)
{
    __shared__ float red[256];
    int t = blockIdx.x, tid = threadIdx.x;
    float rl = rlp ? rlp[idx] : 1.f;
    float xl = (xlp ? xlp[idx] : 0.f) * xls;
    float v[4]; float ss = 0.f;
    #pragma unroll
    for (int i = 0; i < 4; ++i) {
        int c = tid + i * 256;
        float nv = rl * x[(size_t)t * Cx + c] + xl * x0[(size_t)t * Cx + c];
        v[i] = nv; ss += nv * nv;
    }
    red[tid] = ss; __syncthreads();
    for (int s = 128; s > 0; s >>= 1) { if (tid < s) red[tid] += red[tid + s]; __syncthreads(); }
    float inv = rsqrtf(red[0] / Cx + 1e-6f);
    #pragma unroll
    for (int i = 0; i < 4; ++i) {
        int c = tid + i * 256;
        x[(size_t)t * Cx + c]  = v[i];
        xn[(size_t)t * Cx + c] = v[i] * inv;
    }
}

// per (t, head): rope then rms over HD, times scale
__global__ __launch_bounds__(64) void rope_rms_kernel(
    const float* __restrict__ src, float* __restrict__ dst,
    const float* __restrict__ cosb, const float* __restrict__ sinb,
    int ldsrc, int lddst, float scale)
{
    __shared__ float red[64];
    int t = blockIdx.x, h = blockIdx.y, d = threadIdx.x;
    const float* sp = src + (size_t)t * ldsrc + h * HD;
    int dl = d & 31;
    float cs = cosb[t * 32 + dl], sn = sinb[t * 32 + dl];
    float x1 = sp[dl], x2 = sp[dl + 32];
    float val = (d < 32) ? (x1 * cs + x2 * sn) : (-x1 * sn + x2 * cs);
    red[d] = val * val;
    __syncthreads();
    for (int s = 32; s > 0; s >>= 1) { if (d < s) red[d] += red[d + s]; __syncthreads(); }
    float inv = rsqrtf(red[0] / HD + 1e-6f);
    dst[(size_t)t * lddst + h * HD + d] = val * inv * scale;
}

// value-embedding gate: v[t, kvh, :] += 3*sigmoid(xn[t,:12].w[:,kvh]) * ve[id[t]]
__global__ __launch_bounds__(256) void ve_kernel(
    const float* __restrict__ xn, const int* __restrict__ ids,
    const float* __restrict__ vet, const float* __restrict__ vgw,
    float* __restrict__ vout)
{
    int t = blockIdx.x, tid = threadIdx.x;
    int kvh = tid >> 6;
    float dot = 0.f;
    #pragma unroll
    for (int j = 0; j < 12; ++j) dot += xn[(size_t)t * Cx + j] * vgw[j * NKV + kvh];
    float g = 3.f / (1.f + __expf(-dot));
    int id = ids[t];
    vout[(size_t)t * KVx + tid] += g * vet[(size_t)id * KVx + tid];
}

// =====================================================================
// Orchestration
// =====================================================================
extern "C" void kernel_launch(void* const* d_in, const int* in_sizes, int n_in,
                              void* d_out, int out_size, void* d_ws, size_t ws_size,
                              hipStream_t stream)
{
    (void)in_sizes; (void)n_in; (void)out_size; (void)ws_size;

    const int*   ids  = (const int*)d_in[0];
    const float* cosb = (const float*)d_in[1];
    const float* sinb = (const float*)d_in[2];
    const float* wte  = (const float*)d_in[6];
    const float* lmw  = (const float*)d_in[7];
    const float* sgw  = (const float*)d_in[8];
    const float* slam = (const float*)d_in[9];
    const float* rlam = (const float*)d_in[10];
    const float* xlam = (const float*)d_in[11];
    const float* blam = (const float*)d_in[12];
    const float* Wq   = (const float*)d_in[13];
    const float* Wk   = (const float*)d_in[14];
    const float* Wv   = (const float*)d_in[15];
    const float* Wo   = (const float*)d_in[16];
    const float* Wfc  = (const float*)d_in[17];
    const float* Wpr  = (const float*)d_in[18];
    const float* vet  = (const float*)d_in[19];
    const float* vgw  = (const float*)d_in[20];

    float* out      = (float*)d_out;
    float* out_prev = out + (size_t)Tx * VOCAB;
    float* outK     = out_prev + Cx;                       // (L, T, KV)
    float* outV     = outK + (size_t)Lx * Tx * KVx;        // (L, T, KV)

    float* ws   = (float*)d_ws;
    float* xe   = ws;
    float* x    = xe   + (size_t)Tx * Cx;
    float* x0   = x    + (size_t)Tx * Cx;
    float* xn   = x0   + (size_t)Tx * Cx;
    float* xb   = xn   + (size_t)Tx * Cx;
    float* qraw = xb   + (size_t)Tx * Cx;
    float* qhat = qraw + (size_t)Tx * Cx;
    float* ybuf = qhat + (size_t)Tx * Cx;
    float* kraw = ybuf + (size_t)Tx * Cx;   // T*KV
    float* hbuf = kraw + (size_t)Tx * KVx;  // T*FF

    const float qscale = 1.2f * 0.125f;   // 1.2 * HD^-0.5 folded into q
    const float kscale = 1.2f;

    embed_kernel<<<Tx, 256, 0, stream>>>(ids, wte, xe, out_prev);
    smear_kernel<<<Tx, 256, 0, stream>>>(xe, sgw, slam, x, x0);

    for (int i = 0; i < Lx; ++i) {
        float* lK = outK + (size_t)i * Tx * KVx;
        float* lV = outV + (size_t)i * Tx * KVx;

        resid_rms_kernel<<<Tx, 256, 0, stream>>>(x, x0, xn, rlam, xlam, i, 1.0f);

        gemm_bf16_kernel<<<dim3(Cx / GBN, Tx / GBM), 256, 0, stream>>>(
            xn, Wq + (size_t)i * Cx * Cx, qraw, Tx, Cx, Cx, Cx, Cx, 0);
        gemm_bf16_kernel<<<dim3(KVx / GBN, Tx / GBM), 256, 0, stream>>>(
            xn, Wk + (size_t)i * Cx * KVx, kraw, Tx, KVx, Cx, KVx, KVx, 0);
        gemm_bf16_kernel<<<dim3(KVx / GBN, Tx / GBM), 256, 0, stream>>>(
            xn, Wv + (size_t)i * Cx * KVx, lV, Tx, KVx, Cx, KVx, KVx, 0);

        if (i == 0 || i == Lx - 1) {
            int vidx = (i == 0) ? 0 : 1;
            ve_kernel<<<Tx, 256, 0, stream>>>(
                xn, ids, vet + (size_t)vidx * VPAD * KVx, vgw + (size_t)vidx * 12 * NKV, lV);
        }

        rope_rms_kernel<<<dim3(Tx, NH), 64, 0, stream>>>(qraw, qhat, cosb, sinb, Cx, Cx, qscale);
        rope_rms_kernel<<<dim3(Tx, NKV), 64, 0, stream>>>(kraw, lK, cosb, sinb, KVx, KVx, kscale);

        attn_kernel<<<dim3(NH, Tx / 128), 256, 0, stream>>>(qhat, lK, lV, ybuf);

        gemm_bf16_kernel<<<dim3(Cx / GBN, Tx / GBM), 256, 0, stream>>>(
            ybuf, Wo + (size_t)i * Cx * Cx, x, Tx, Cx, Cx, Cx, Cx, 1 /*add*/);

        resid_rms_kernel<<<Tx, 256, 0, stream>>>(x, x, xn, nullptr, nullptr, 0, 0.f);

        gemm_bf16_kernel<<<dim3(FFx / GBN, Tx / GBM), 256, 0, stream>>>(
            xn, Wfc + (size_t)i * Cx * FFx, hbuf, Tx, FFx, Cx, FFx, FFx, 2 /*relu^2*/);
        gemm_bf16_kernel<<<dim3(Cx / GBN, Tx / GBM), 256, 0, stream>>>(
            hbuf, Wpr + (size_t)i * FFx * Cx, x, Tx, Cx, FFx, Cx, Cx, 1 /*add*/);

        if (i == Lx / 2) {
            hipMemcpyAsync(xb, x, (size_t)Tx * Cx * sizeof(float),
                           hipMemcpyDeviceToDevice, stream);
        }
    }

    // x = x - backout_lambda * xb ; xn = rms(x)
    resid_rms_kernel<<<Tx, 256, 0, stream>>>(x, xb, xn, nullptr, blam, 0, -1.0f);

    // logits = 15*tanh((xn @ lm_head)/15), cols < VOCAB, ldc = VOCAB
    gemm_bf16_kernel<<<dim3(VPAD / GBN, Tx / GBM), 256, 0, stream>>>(
        xn, lmw, out, Tx, VPAD, Cx, VOCAB, VOCAB, 3 /*tanh15*/);
}
